// L2_CRGNN_6030134084106
// MI455X (gfx1250) — 8-bit matrix operand order repaired and checked by reading, compiled, not run on hardware
//
#include <hip/hip_runtime.h>

#define N 4096

typedef __attribute__((ext_vector_type(8))) int      v8i;
typedef __attribute__((ext_vector_type(8))) float    v8f;
typedef __attribute__((ext_vector_type(2))) float    v2f;
typedef __attribute__((ext_vector_type(4))) unsigned v4u;

__global__ void k_prep(const float* __restrict__ adj, signed char* __restrict__ adj8,
                       unsigned* __restrict__ bits, int* __restrict__ deg,
                       unsigned short* __restrict__ nbr, int* __restrict__ ncnt) {
  int v = blockIdx.x, t = threadIdx.x;
  __shared__ unsigned wb[128];
  const float* row = adj + (size_t)v * N;
  unsigned word = 0;
  for (int q = 0; q < 8; ++q) {
    unsigned pv = 0;
    for (int b = 0; b < 4; ++b) {
      int j = t * 32 + q * 4 + b;
      unsigned bit = row[j] > 0.5f ? 1u : 0u;
      pv |= bit << (8 * b);
      word |= bit << (q * 4 + b);
    }
    ((unsigned*)adj8)[(size_t)v * (N / 4) + t * 8 + q] = pv;
  }
  wb[t] = word;
  bits[(size_t)v * 128 + t] = word;
  __syncthreads();
  if (t == 0) {
    int d = 0;
    for (int i = 0; i < 128; ++i) d += __popc(wb[i]);
    deg[v] = d;
    int cnt = 0;
    for (int w = 0; w < 128 && cnt < 64; ++w) {
      unsigned x = wb[w];
      if (w == (v >> 5)) x |= 1u << (v & 31);
      while (x && cnt < 64) {
        int b = __ffs(x) - 1; x &= x - 1;
        nbr[(size_t)v * 64 + cnt] = (unsigned short)(w * 32 + b);
        ++cnt;
      }
    }
    ncnt[v] = cnt;
  }
}

__global__ void k_gemm_iu8(const signed char* __restrict__ M8,
                           signed char* __restrict__ C8, int* __restrict__ C32,
                           int mode) {
  __shared__ v4u At4[2][256];
  __shared__ v4u Bt4[2][256];

  int t = threadIdx.x;
  int wave = t >> 5, lane = t & 31;
  int r0 = blockIdx.y * 64, c0 = blockIdx.x * 64;
  int hi = lane >> 4;

  int row = t >> 1;
  int cb  = (t & 1) * 32;

  const char* pa = (const char*)(M8 + (size_t)(r0 + row) * N + cb);
  const char* pb = (const char*)(M8 + (size_t)(c0 + row) * N + cb);
  v4u ra0 = *(const v4u*)(pa);      v4u ra1 = *(const v4u*)(pa + 16);
  v4u rb0 = *(const v4u*)(pb);      v4u rb1 = *(const v4u*)(pb + 16);
  At4[0][2 * t] = ra0; At4[0][2 * t + 1] = ra1;
  Bt4[0][2 * t] = rb0; Bt4[0][2 * t + 1] = rb1;
  __syncthreads();

  v8i acc[4] = {};
  int cur = 0;
  for (int k0 = 0; k0 < N; k0 += 64) {
    bool more = (k0 + 64) < N;
    if (more) {
      const char* qa = (const char*)(M8 + (size_t)(r0 + row) * N + k0 + 64 + cb);
      const char* qb = (const char*)(M8 + (size_t)(c0 + row) * N + k0 + 64 + cb);
      ra0 = *(const v4u*)(qa); ra1 = *(const v4u*)(qa + 16);
      rb0 = *(const v4u*)(qb); rb1 = *(const v4u*)(qb + 16);
      if (k0 + 128 < N && (t & 1) == 0) {
        __builtin_prefetch(M8 + (size_t)(r0 + row) * N + k0 + 128, 0, 0);
        __builtin_prefetch(M8 + (size_t)(c0 + row) * N + k0 + 128, 0, 0);
      }
    }

    const unsigned char* At = (const unsigned char*)At4[cur];
    const unsigned char* Bt = (const unsigned char*)Bt4[cur];

    const unsigned char* ap = At + (wave * 16 + (lane & 15)) * 64 + (hi << 3);
    v8i a;
    a[0] = *(const int*)(ap + 0);  a[1] = *(const int*)(ap + 4);
    a[2] = *(const int*)(ap + 16); a[3] = *(const int*)(ap + 20);
    a[4] = *(const int*)(ap + 32); a[5] = *(const int*)(ap + 36);
    a[6] = *(const int*)(ap + 48); a[7] = *(const int*)(ap + 52);

#pragma unroll
    for (int cs = 0; cs < 4; ++cs) {
      const unsigned char* bp = Bt + (cs * 16 + (lane & 15)) * 64 + (hi << 3);
      v8i b;
      b[0] = *(const int*)(bp + 0);  b[1] = *(const int*)(bp + 4);
      b[2] = *(const int*)(bp + 16); b[3] = *(const int*)(bp + 20);
      b[4] = *(const int*)(bp + 32); b[5] = *(const int*)(bp + 36);
      b[6] = *(const int*)(bp + 48); b[7] = *(const int*)(bp + 52);
      acc[cs] = __builtin_amdgcn_wmma_i32_16x16x64_iu8(false, a, false, b,
                                                       acc[cs], false, false);
    }

    if (more) {
      int nxt = cur ^ 1;
      At4[nxt][2 * t] = ra0; At4[nxt][2 * t + 1] = ra1;
      Bt4[nxt][2 * t] = rb0; Bt4[nxt][2 * t + 1] = rb1;
      __syncthreads();
      cur = nxt;
    }
  }

#pragma unroll
  for (int cs = 0; cs < 4; ++cs) {
#pragma unroll
    for (int g = 0; g < 8; ++g) {
      int rr = r0 + wave * 16 + g + (hi << 3);
      int cc = c0 + cs * 16 + (lane & 15);
      int val = acc[cs][g];
      if (mode == 0) C8[(size_t)rr * N + cc] = (signed char)(val > 127 ? 127 : val);
      else           C32[(size_t)rr * N + cc] = val;
    }
  }
}

__global__ void k_ew(const signed char* __restrict__ a28, const unsigned* __restrict__ bits,
                     float* __restrict__ Ebuf, float* __restrict__ Wbuf) {
  int v = blockIdx.x, t = threadIdx.x;
  __shared__ int sE[128], sW[128];
  int e = 0, w = 0;
  for (int j = t; j < N; j += 128) {
    int aij = (bits[(size_t)v * 128 + (j >> 5)] >> (j & 31)) & 1;
    int m = aij + (j == v);
    if (m) {
      int D = (int)a28[(size_t)v * N + j] + aij;
      e += D;
      w += D * (D - 1);
    }
  }
  sE[t] = e; sW[t] = w;
  __syncthreads();
  for (int st = 64; st > 0; st >>= 1) {
    if (t < st) { sE[t] += sE[t + st]; sW[t] += sW[t + st]; }
    __syncthreads();
  }
  if (t == 0) { Ebuf[v] = 0.5f * (float)sE[0]; Wbuf[v] = 0.5f * (float)sW[0]; }
}

__global__ void k_tri(const unsigned* __restrict__ bits, const unsigned short* __restrict__ nbr,
                      const int* __restrict__ ncnt, const int* __restrict__ deg,
                      const float* __restrict__ Ebuf, const float* __restrict__ Wbuf,
                      float* __restrict__ feat) {
  int v = blockIdx.x, t = threadIdx.x;
  __shared__ unsigned short L[64];
  __shared__ unsigned long long rows[64];
  __shared__ int part[64];
  int cnt = ncnt[v];
  L[t] = (t < cnt) ? nbr[(size_t)v * 64 + t] : (unsigned short)0;
  __syncthreads();
  unsigned long long r = 0ull;
  if (t < cnt) {
    const unsigned* brow = bits + (size_t)L[t] * 128;
    for (int j = 0; j < cnt; ++j) {
      int lj = L[j];
      unsigned long long bit = (brow[lj >> 5] >> (lj & 31)) & 1u;
      r |= bit << j;
    }
  }
  rows[t] = r;
  __syncthreads();
  int s = 0;
  unsigned long long rr = r;
  while (rr) {
    int j = __builtin_ctzll(rr); rr &= rr - 1;
    s += __builtin_popcountll(r & rows[j]);
  }
  part[t] = s;
  __syncthreads();
  for (int st = 32; st > 0; st >>= 1) {
    if (t < st) part[t] += part[t + st];
    __syncthreads();
  }
  if (t == 0) {
    float T = (float)part[0] / 6.0f;
    float k = (float)deg[v] + 1.0f;
    float E = Ebuf[v], W = Wbuf[v];
    float f3 = T;
    float f2 = W - 3.0f * T;
    float f1 = E * (k - 2.0f) - 2.0f * f2 - 3.0f * f3;
    float tot = k * (k - 1.0f) * (k - 2.0f) / 6.0f;
    float f0 = tot - f1 - f2 - f3;
    if (k < 3.0f) { f0 = f1 = f2 = f3 = 0.0f; }
    float sinv = 1.0f / (f0 + f1 + f2 + f3 + 1e-10f);
    float* fp = feat + (size_t)v * 15;
    fp[0] = f0 * sinv; fp[1] = f1 * sinv; fp[2] = f2 * sinv; fp[3] = f3 * sinv;
  }
}

__global__ void k_top8(const int* __restrict__ a4, const int* __restrict__ deg,
                       float* __restrict__ feat) {
  int v = blockIdx.x, t = threadIdx.x;
  __shared__ int tops[128 * 8];
  int best[8];
#pragma unroll
  for (int q = 0; q < 8; ++q) best[q] = -2147483647 - 1;
  for (int j = t; j < N; j += 128) {
    int val = a4[(size_t)v * N + j];
    if (val > best[7]) {
      int p = 7;
      while (p > 0 && best[p - 1] < val) { best[p] = best[p - 1]; --p; }
      best[p] = val;
    }
  }
#pragma unroll
  for (int q = 0; q < 8; ++q) tops[t * 8 + q] = best[q];
  __syncthreads();
  if (t == 0) {
    int b[8];
#pragma unroll
    for (int q = 0; q < 8; ++q) b[q] = -2147483647 - 1;
    for (int i = 0; i < 128 * 8; ++i) {
      int val = tops[i];
      if (val > b[7]) {
        int p = 7;
        while (p > 0 && b[p - 1] < val) { b[p] = b[p - 1]; --p; }
        b[p] = val;
      }
    }
    float* fp = feat + (size_t)v * 15;
#pragma unroll
    for (int q = 0; q < 8; ++q) fp[4 + q] = (float)b[q];
    float d = (float)deg[v];
    fp[12] = d; fp[13] = d * d; fp[14] = d;
  }
}

__global__ void k_enc(const float* __restrict__ feat, const float* __restrict__ e_w,
                      const float* __restrict__ e_b, float* __restrict__ x) {
  int idx = blockIdx.x * 256 + threadIdx.x;
  int v = idx >> 6, c = idx & 63;
  float s = e_b[c];
#pragma unroll
  for (int f = 0; f < 15; ++f) s += feat[(size_t)v * 15 + f] * e_w[f * 64 + c];
  x[idx] = s;
}

__global__ void k_agg(const float* __restrict__ x, float* __restrict__ h,
                      const unsigned short* __restrict__ nbr, const int* __restrict__ ncnt,
                      const float* __restrict__ eps, int layer) {
  int v = blockIdx.x, c = threadIdx.x;
  __shared__ unsigned short L[64];
  __shared__ int scnt;
  L[c] = nbr[(size_t)v * 64 + c];
  if (c == 0) scnt = ncnt[v];
  __syncthreads();
  int cnt = scnt;
  float acc = (1.0f + eps[layer]) * x[(size_t)v * 64 + c];
  for (int i = 0; i < cnt; ++i) {
    int j = L[i];
    if (j != v) acc += x[(size_t)j * 64 + c];
  }
  h[(size_t)v * 64 + c] = acc;
}

__global__ void k_mlp(const float* __restrict__ h, float* __restrict__ x,
                      const float* __restrict__ w1, const float* __restrict__ b1,
                      const float* __restrict__ w2, const float* __restrict__ b2) {
  __shared__ float hs[64 * 64];
  __shared__ float wsm[64 * 64];
  __shared__ float ts[64 * 64];
  int t = threadIdx.x;
  int wave = t >> 5, lane = t & 31;
  int hi = lane >> 4;
  int r0 = blockIdx.x * 64;

  for (int i = t; i < 4096; i += 128) {
    hs[i] = h[(size_t)r0 * 64 + i];
    wsm[i] = w1[i];
  }
  __syncthreads();

  v8f acc[4] = {};
  for (int k0 = 0; k0 < 64; k0 += 4) {
    v2f a;
    a[0] = hs[(wave * 16 + (lane & 15)) * 64 + k0 + 2 * hi];
    a[1] = hs[(wave * 16 + (lane & 15)) * 64 + k0 + 2 * hi + 1];
#pragma unroll
    for (int cs = 0; cs < 4; ++cs) {
      v2f b;
      b[0] = wsm[(k0 + 2 * hi) * 64 + cs * 16 + (lane & 15)];
      b[1] = wsm[(k0 + 2 * hi + 1) * 64 + cs * 16 + (lane & 15)];
      acc[cs] = __builtin_amdgcn_wmma_f32_16x16x4_f32(false, a, false, b,
                                                      (short)0, acc[cs], false, false);
    }
  }
#pragma unroll
  for (int cs = 0; cs < 4; ++cs) {
#pragma unroll
    for (int g = 0; g < 8; ++g) {
      int rr = wave * 16 + g + (hi << 3);
      int cc = cs * 16 + (lane & 15);
      float val = acc[cs][g] + b1[cc];
      ts[rr * 64 + cc] = val > 0.0f ? val : 0.0f;
    }
  }
  __syncthreads();
  for (int i = t; i < 4096; i += 128) wsm[i] = w2[i];
  __syncthreads();

  v8f acc2[4] = {};
  for (int k0 = 0; k0 < 64; k0 += 4) {
    v2f a;
    a[0] = ts[(wave * 16 + (lane & 15)) * 64 + k0 + 2 * hi];
    a[1] = ts[(wave * 16 + (lane & 15)) * 64 + k0 + 2 * hi + 1];
#pragma unroll
    for (int cs = 0; cs < 4; ++cs) {
      v2f b;
      b[0] = wsm[(k0 + 2 * hi) * 64 + cs * 16 + (lane & 15)];
      b[1] = wsm[(k0 + 2 * hi + 1) * 64 + cs * 16 + (lane & 15)];
      acc2[cs] = __builtin_amdgcn_wmma_f32_16x16x4_f32(false, a, false, b,
                                                       (short)0, acc2[cs], false, false);
    }
  }
#pragma unroll
  for (int cs = 0; cs < 4; ++cs) {
#pragma unroll
    for (int g = 0; g < 8; ++g) {
      int rr = wave * 16 + g + (hi << 3);
      int cc = cs * 16 + (lane & 15);
      x[(size_t)(r0 + rr) * 64 + cc] = acc2[cs][g] + b2[cc];
    }
  }
}

__global__ void k_sum(const float* __restrict__ x, float* __restrict__ out) {
  int c = blockIdx.x, t = threadIdx.x;
  __shared__ float s[256];
  float a = 0.0f;
  for (int v = t; v < N; v += 256) a += x[(size_t)v * 64 + c];
  s[t] = a;
  __syncthreads();
  for (int st = 128; st > 0; st >>= 1) {
    if (t < st) s[t] += s[t + st];
    __syncthreads();
  }
  if (t == 0) out[c] = s[0];
}

extern "C" void kernel_launch(void* const* d_in, const int* in_sizes, int n_in,
                              void* d_out, int out_size, void* d_ws, size_t ws_size,
                              hipStream_t stream) {
  const float* adj = (const float*)d_in[0];
  const float* e_w = (const float*)d_in[1];
  const float* e_b = (const float*)d_in[2];
  const float* w1  = (const float*)d_in[3];
  const float* b1  = (const float*)d_in[4];
  const float* w2  = (const float*)d_in[5];
  const float* b2  = (const float*)d_in[6];
  const float* eps = (const float*)d_in[7];
  float* out = (float*)d_out;

  char* wsb = (char*)d_ws;
  signed char*    adj8 = (signed char*)(wsb + 0);
  signed char*    a28  = (signed char*)(wsb + 16777216);
  int*            a4   = (int*)(wsb + 33554432);
  unsigned*       bits = (unsigned*)(wsb + 100663296);
  int*            deg  = (int*)(wsb + 102760448);
  int*            ncnt = (int*)(wsb + 102776832);
  unsigned short* nbr  = (unsigned short*)(wsb + 102793216);
  float*          Ebuf = (float*)(wsb + 103317504);
  float*          Wbuf = (float*)(wsb + 103333888);
  float*          feat = (float*)(wsb + 103350272);
  float*          x    = (float*)(wsb + 103596032);
  float*          h    = (float*)(wsb + 104644608);

  k_prep<<<N, 128, 0, stream>>>(adj, adj8, bits, deg, nbr, ncnt);
  k_gemm_iu8<<<dim3(64, 64), 128, 0, stream>>>(adj8, a28, nullptr, 0);
  k_gemm_iu8<<<dim3(64, 64), 128, 0, stream>>>(a28, nullptr, a4, 1);
  k_ew<<<N, 128, 0, stream>>>(a28, bits, Ebuf, Wbuf);
  k_tri<<<N, 64, 0, stream>>>(bits, nbr, ncnt, deg, Ebuf, Wbuf, feat);
  k_top8<<<N, 128, 0, stream>>>(a4, deg, feat);
  k_enc<<<(N * 64) / 256, 256, 0, stream>>>(feat, e_w, e_b, x);
  for (int l = 0; l < 3; ++l) {
    k_agg<<<N, 64, 0, stream>>>(x, h, nbr, ncnt, eps, l);
    k_mlp<<<N / 64, 128, 0, stream>>>(h, x, w1 + l * 4096, b1 + l * 64,
                                      w2 + l * 4096, b2 + l * 64);
  }
  k_sum<<<64, 256, 0, stream>>>(x, out);
}
